// CloudCropV2AttDeA_27702539059798
// MI455X (gfx1250) — matrix operand order repaired and checked by reading, not compiled, not run on hardware
//
#include <hip/hip_runtime.h>
#include <math.h>

typedef __bf16 bf16;
typedef bf16  v16bf __attribute__((ext_vector_type(16)));
typedef bf16  v8bf  __attribute__((ext_vector_type(8)));
typedef float v8f   __attribute__((ext_vector_type(8)));

// ---------------- WMMA helpers (gfx1250, wave32) ----------------
__device__ __forceinline__ v8f wmma_bf16(v16bf a, v16bf b, v8f c) {
  return __builtin_amdgcn_wmma_f32_16x16x32_bf16(false, a, false, b, (short)0, c,
                                                 false, false);
}

// A fragment (16x32 bf16): lane holds row (lane&15); half=lane>>4.
// element e: K = half*8 + e (e<8) ; K = 16 + half*8 + (e-8) (e>=8)
__device__ __forceinline__ v16bf frag_a(const bf16* rowPtr, int half) {
  v8bf lo = *(const v8bf*)(rowPtr + half * 8);
  v8bf hi = *(const v8bf*)(rowPtr + 16 + half * 8);
  v16bf r;
#pragma unroll
  for (int i = 0; i < 8; i++) { r[i] = lo[i]; r[8 + i] = hi[i]; }
  return r;
}
// B fragment (32x16 bf16): lane holds col (lane&15); element e: K = half*8 + e (e<8) ; K = 16 + half*8 + (e-8) (e>=8)
// colPtr points at K-contiguous storage for this column.
__device__ __forceinline__ v16bf frag_b(const bf16* colPtr, int half) {
  v8bf lo = *(const v8bf*)(colPtr + half * 8);
  v8bf hi = *(const v8bf*)(colPtr + 16 + half * 8);
  v16bf r;
#pragma unroll
  for (int i = 0; i < 8; i++) { r[i] = lo[i]; r[8 + i] = hi[i]; }
  return r;
}
__device__ __forceinline__ v8f vzero8() {
  v8f z;
#pragma unroll
  for (int i = 0; i < 8; i++) z[i] = 0.f;
  return z;
}

// ---------------- problem constants ----------------
#define NPTS 512
#define TGRP 64            // 16 grouped + 48 region points
#define M1COLS 65536       // 2*512*64
#define M2COLS 786432      // 24*512*64
#define K1PAD 544          // 515 -> 544
#define CPAD 96            // 67 -> 96 (attention / f1)

// ---------------- workspace offsets (bytes, all 16B aligned) ----------------
#define OFF_IDX   0u               // 1024*16 int            = 65536
#define OFF_XYZ   65536u           // 2*3*512*64 f32         = 786432
#define OFF_S1    851968u          // 256 f32
#define OFF_SH1   852992u
#define OFF_S2    854016u          // 768 f32
#define OFF_SH2   857088u
#define OFF_SF1   860160u          // 128 f32
#define OFF_SHF1  860672u
#define OFF_SF2   861184u
#define OFF_SHF2  861696u
#define OFF_WQ    862208u          // 96*96 bf16 = 18432
#define OFF_WK    880640u
#define OFF_WV    899072u
#define OFF_WO    917504u
#define OFF_W1P   935936u          // 256*544 bf16 = 278528
#define OFF_W2P   1214464u         // 768*256 bf16 = 393216
#define OFF_F1P   1607680u         // 128*96 bf16
#define OFF_F2P   1632256u         // 128*128 bf16
#define OFF_H1    1665024u         // 256*65536 bf16 = 33554432
#define OFF_H2    35219456u        // 768*65536 bf16 = 100663296
#define OFF_X1    135882752u       // 544*65536 bf16 (dead before Y written)
#define OFF_Y     135882752u       // 96*786432 bf16 = 150994944 ; end ~287MB

// ================= weight / BN packing =================
__global__ void pack_w_kernel(const float* __restrict__ w, bf16* __restrict__ out,
                              int Oi, int Ci, int Op, int Cp) {
  int i = blockIdx.x * 256 + threadIdx.x;
  if (i >= Op * Cp) return;
  int o = i / Cp, c = i % Cp;
  out[i] = (o < Oi && c < Ci) ? (bf16)w[o * Ci + c] : (bf16)0.f;
}

__global__ void pack_bn_kernel(const float* __restrict__ b, const float* __restrict__ g,
                               const float* __restrict__ be, const float* __restrict__ mu,
                               const float* __restrict__ v, float* __restrict__ sc,
                               float* __restrict__ sh, int C) {
  int i = blockIdx.x * 256 + threadIdx.x;
  if (i >= C) return;
  float s = g[i] * rsqrtf(v[i] + 1e-5f);
  sc[i] = s;
  sh[i] = be[i] + s * (b[i] - mu[i]);
}

// ================= cylinder query =================
__global__ void cylinder_query_kernel(const float* __restrict__ gp_f,
                                      const float* __restrict__ rot,
                                      int* __restrict__ idx) {
  int q = blockIdx.x * 256 + threadIdx.x;
  if (q >= 2 * NPTS) return;
  int b = q >> 9, n = q & 511;
  float cx = gp_f[(b * 3 + 0) * NPTS + n];
  float cy = gp_f[(b * 3 + 1) * NPTS + n];
  float cz = gp_f[(b * 3 + 2) * NPTS + n];
  const float* R = rot + (size_t)q * 9;
  float r00 = R[0], r01 = R[1], r02 = R[2];
  float r10 = R[3], r11 = R[4], r12 = R[5];
  float r20 = R[6], r21 = R[7], r22 = R[8];
  int out[16];
  int cnt = 0;
  for (int k = 0; k < NPTS && cnt < 16; k++) {
    float dx = gp_f[(b * 3 + 0) * NPTS + k] - cx;
    float dy = gp_f[(b * 3 + 1) * NPTS + k] - cy;
    float dz = gp_f[(b * 3 + 2) * NPTS + k] - cz;
    float a0 = dx * r00 + dy * r10 + dz * r20;
    float a1 = dx * r01 + dy * r11 + dz * r21;
    float a2 = dx * r02 + dy * r12 + dz * r22;
    bool valid = (a1 * a1 + a2 * a2 < 0.0025f) && (a0 > -0.02f) && (a0 < 0.04f);
    if (valid) out[cnt++] = k;
  }
  int first = (cnt > 0) ? out[0] : NPTS;
  for (int s = cnt; s < 16; s++) out[s] = first;
  for (int s = 0; s < 16; s++) idx[q * 16 + s] = out[s];
}

// ================= build padded activation matrix X1 [544, 65536] =================
__global__ __launch_bounds__(256) void build_feats_kernel(
    const float* __restrict__ gp_f, const float* __restrict__ gp_feats,
    const float* __restrict__ rp_f, const float* __restrict__ rp_feats,
    const float* __restrict__ rot, const int* __restrict__ idx,
    bf16* __restrict__ X1, float* __restrict__ xyz) {
  __shared__ int keys[16];
  int g = blockIdx.x;            // b*512 + n
  int b = g >> 9, n = g & 511;
  int tid = threadIdx.x;
  if (tid < 16) keys[tid] = idx[g * 16 + tid];
  __syncthreads();
  if (tid < TGRP) {
    int t = tid;
    float cx = gp_f[(b * 3 + 0) * NPTS + n];
    float cy = gp_f[(b * 3 + 1) * NPTS + n];
    float cz = gp_f[(b * 3 + 2) * NPTS + n];
    const float* R = rot + (size_t)g * 9;
    float px, py, pz;
    if (t < 16) {
      int k = keys[t];
      px = gp_f[(b * 3 + 0) * NPTS + k];
      py = gp_f[(b * 3 + 1) * NPTS + k];
      pz = gp_f[(b * 3 + 2) * NPTS + k];
    } else {
      int rr = t - 16;
      px = rp_f[((size_t)(b * 3 + 0) * NPTS + n) * 48 + rr];
      py = rp_f[((size_t)(b * 3 + 1) * NPTS + n) * 48 + rr];
      pz = rp_f[((size_t)(b * 3 + 2) * NPTS + n) * 48 + rr];
    }
    float dx = (px - cx) * 20.f, dy = (py - cy) * 20.f, dz = (pz - cz) * 20.f;
    float a0 = dx * R[0] + dy * R[3] + dz * R[6];
    float a1 = dx * R[1] + dy * R[4] + dz * R[7];
    float a2 = dx * R[2] + dy * R[5] + dz * R[8];
    size_t m = (size_t)g * TGRP + t;
    xyz[((size_t)(b * 3 + 0) * NPTS + n) * TGRP + t] = a0;
    xyz[((size_t)(b * 3 + 1) * NPTS + n) * TGRP + t] = a1;
    xyz[((size_t)(b * 3 + 2) * NPTS + n) * TGRP + t] = a2;
    X1[(size_t)0 * M1COLS + m] = (bf16)a0;
    X1[(size_t)1 * M1COLS + m] = (bf16)a1;
    X1[(size_t)2 * M1COLS + m] = (bf16)a2;
  }
  __syncthreads();
  for (int i = tid; i < 512 * TGRP; i += 256) {
    int c = i >> 6, t = i & 63;
    float v;
    if (t < 16)
      v = gp_feats[((size_t)(b * 512 + c)) * NPTS + keys[t]];
    else
      v = rp_feats[(((size_t)(b * 512 + c)) * NPTS + n) * 48 + (t - 16)];
    X1[(size_t)(3 + c) * M1COLS + (size_t)g * TGRP + t] = (bf16)v;
  }
  for (int i = tid; i < 29 * TGRP; i += 256) {
    int c = 515 + (i >> 6), t = i & 63;
    X1[(size_t)c * M1COLS + (size_t)g * TGRP + t] = (bf16)0.f;
  }
}

// ================= generic 128x128-tile bf16 WMMA GEMM + BN + ReLU =================
// Out[o, m] = relu(scale[o] * (sum_k W[o,k] X[k,m]) + shift[o]); stored bf16.
// Double-buffered LDS tiles: stage k-tile (t+1) while WMMAs consume tile t.
__global__ __launch_bounds__(256) void gemm_bn_relu_kernel(
    const bf16* __restrict__ W, const bf16* __restrict__ X, bf16* __restrict__ Out,
    const float* __restrict__ scale, const float* __restrict__ shift, int M, int Kp) {
  __shared__ alignas(16) bf16 As[2][128 * 40];
  __shared__ alignas(16) bf16 Bs[2][128 * 40];   // transposed: [col][k]
  const int tid = threadIdx.x;
  const int lane = tid & 31, wv = tid >> 5;
  const int half = (lane >> 4) & 1, l16 = lane & 15;
  const int wr = wv >> 1, wc = wv & 1;
  const int tileO = blockIdx.y * 128, tileM = blockIdx.x * 128;
  v8f acc[2][4];
#pragma unroll
  for (int a = 0; a < 2; a++)
#pragma unroll
    for (int c = 0; c < 4; c++) acc[a][c] = vzero8();

  auto stage = [&](int buf, int k0) {
    {  // stage A tile [128 rows][32 k]
      int row = tid >> 1, ko = (tid & 1) * 16;
      const bf16* src = W + (size_t)(tileO + row) * Kp + k0 + ko;
      *(v8bf*)(&As[buf][row * 40 + ko]) = *(const v8bf*)(src);
      *(v8bf*)(&As[buf][row * 40 + ko + 8]) = *(const v8bf*)(src + 8);
    }
    {  // stage B tile transposed: [128 cols][32 k]
      int kk = tid >> 3, c0 = (tid & 7) * 16;
      const bf16* src = X + (size_t)(k0 + kk) * M + tileM + c0;
      v8bf v0 = *(const v8bf*)(src);
      v8bf v1 = *(const v8bf*)(src + 8);
#pragma unroll
      for (int j = 0; j < 8; j++) {
        Bs[buf][(c0 + j) * 40 + kk] = v0[j];
        Bs[buf][(c0 + 8 + j) * 40 + kk] = v1[j];
      }
    }
  };

  const int nk = Kp >> 5;
  stage(0, 0);
  __syncthreads();
  for (int kt = 0; kt < nk; kt++) {
    int cur = kt & 1;
    if (kt + 2 < nk) {  // speculative prefetch of the streaming matrix (HBM-resident)
      __builtin_prefetch(X + (size_t)((kt + 2) * 32 + (tid >> 3)) * M + tileM +
                             (tid & 7) * 16,
                         0, 0);
    }
    if (kt + 1 < nk) stage(cur ^ 1, (kt + 1) * 32);
#pragma unroll
    for (int mi = 0; mi < 2; mi++) {
      v16bf a = frag_a(&As[cur][(wr * 32 + mi * 16 + l16) * 40], half);
#pragma unroll
      for (int ni = 0; ni < 4; ni++) {
        v16bf bb = frag_b(&Bs[cur][(wc * 64 + ni * 16 + l16) * 40], half);
        acc[mi][ni] = wmma_bf16(a, bb, acc[mi][ni]);
      }
    }
    __syncthreads();
  }
#pragma unroll
  for (int mi = 0; mi < 2; mi++)
#pragma unroll
    for (int ni = 0; ni < 4; ni++)
#pragma unroll
      for (int j = 0; j < 8; j++) {
        int o = tileO + wr * 32 + mi * 16 + j + half * 8;
        int col = tileM + wc * 64 + ni * 16 + l16;
        float y = scale[o] * acc[mi][ni][j] + shift[o];
        y = y > 0.f ? y : 0.f;
        Out[(size_t)o * M + col] = (bf16)y;
      }
}

// ================= fused attention (per (b*12+head, n) group) =================
__global__ __launch_bounds__(128) void attn_kernel(
    const bf16* __restrict__ H2, const float* __restrict__ xyz,
    const bf16* __restrict__ Wq, const bf16* __restrict__ Wk,
    const bf16* __restrict__ Wv, const bf16* __restrict__ Wo,
    const float* __restrict__ bq, const float* __restrict__ bk,
    const float* __restrict__ bv, const float* __restrict__ bo,
    const float* __restrict__ lng, const float* __restrict__ lnb,
    bf16* __restrict__ Yout) {
  __shared__ alignas(16) char sm[53248];
  bf16* Xp = (bf16*)sm;                 // [64][96]  x (padded), kept for residual
  bf16* Qm = (bf16*)(sm + 12288);       // [64][96]  then P [64][64]
  bf16* Km = (bf16*)(sm + 24576);       // [64][96]  then Vt [96][64]
  float* S = (float*)(sm + 36864);      // [64][64]  then O bf16 [64][96]
  bf16* P = Qm;
  bf16* Vt = Km;
  bf16* Om = (bf16*)(sm + 36864);

  int g = blockIdx.x;  // bh*512 + n
  int bh = g >> 9, n = g & 511;
  int b = bh / 12, head = bh % 12;
  int tid = threadIdx.x, lane = tid & 31, wv = tid >> 5;
  int half = (lane >> 4) & 1, l16 = lane & 15;
  size_t colbase = (size_t)g * TGRP;
  size_t h2colbase = ((size_t)(b * NPTS + n)) * TGRP;

  // ---- load x padded (67 -> 96); t-fastest so lanes read contiguous bf16
  for (int i = tid; i < CPAD * 64; i += 128) {
    int c = i >> 6, t = i & 63;
    float v;
    if (c < 3)
      v = xyz[(((size_t)(b * 3 + c) * NPTS + n) * TGRP) + t];
    else if (c < 67)
      v = (float)H2[(size_t)(head * 64 + (c - 3)) * M1COLS + h2colbase + t];
    else
      v = 0.f;
    Xp[t * CPAD + c] = (bf16)v;
  }
  __syncthreads();

  auto project = [&](const bf16* Wm, const float* bias, bf16* dst, bool transposed) {
    v8f acc[6];
#pragma unroll
    for (int i = 0; i < 6; i++) acc[i] = vzero8();
    for (int kc = 0; kc < 3; kc++) {
      v16bf a = frag_a(Xp + (wv * 16 + l16) * CPAD + kc * 32, half);
#pragma unroll
      for (int nt = 0; nt < 6; nt++) {
        v16bf bb = frag_b(Wm + (size_t)(nt * 16 + l16) * CPAD + kc * 32, half);
        acc[nt] = wmma_bf16(a, bb, acc[nt]);
      }
    }
#pragma unroll
    for (int nt = 0; nt < 6; nt++)
#pragma unroll
      for (int j = 0; j < 8; j++) {
        int t = wv * 16 + j + half * 8;
        int o = nt * 16 + l16;
        float y = acc[nt][j] + (o < 67 ? bias[o] : 0.f);
        if (transposed)
          dst[o * 64 + t] = (bf16)y;
        else
          dst[t * CPAD + o] = (bf16)y;
      }
  };

  project(Wq, bq, Qm, false);
  project(Wk, bk, Km, false);
  __syncthreads();

  // ---- scores S = (Q K^T) / sqrt(67)
  {
    v8f acc[4];
#pragma unroll
    for (int i = 0; i < 4; i++) acc[i] = vzero8();
    for (int kc = 0; kc < 3; kc++) {
      v16bf a = frag_a(Qm + (wv * 16 + l16) * CPAD + kc * 32, half);
#pragma unroll
      for (int ut = 0; ut < 4; ut++) {
        v16bf bb = frag_b(Km + (ut * 16 + l16) * CPAD + kc * 32, half);
        acc[ut] = wmma_bf16(a, bb, acc[ut]);
      }
    }
    const float sc = 0.122169444f;  // 1/sqrt(67)
#pragma unroll
    for (int ut = 0; ut < 4; ut++)
#pragma unroll
      for (int j = 0; j < 8; j++) {
        int t = wv * 16 + j + half * 8;
        int u = ut * 16 + l16;
        S[t * 64 + u] = acc[ut][j] * sc;
      }
  }
  __syncthreads();

  // ---- V projection (into old K space, transposed) + softmax (into old Q space)
  project(Wv, bv, Vt, true);
  if (tid < 64) {
    int t = tid;
    float mx = -1e30f;
    for (int u = 0; u < 64; u++) mx = fmaxf(mx, S[t * 64 + u]);
    float sum = 0.f;
    for (int u = 0; u < 64; u++) sum += __expf(S[t * 64 + u] - mx);
    float inv = 1.f / sum;
    for (int u = 0; u < 64; u++)
      P[t * 64 + u] = (bf16)(__expf(S[t * 64 + u] - mx) * inv);
  }
  __syncthreads();

  // ---- O = P @ V (into old S space)
  {
    v8f acc[6];
#pragma unroll
    for (int i = 0; i < 6; i++) acc[i] = vzero8();
    for (int kc = 0; kc < 2; kc++) {
      v16bf a = frag_a(P + (wv * 16 + l16) * 64 + kc * 32, half);
#pragma unroll
      for (int nt = 0; nt < 6; nt++) {
        v16bf bb = frag_b(Vt + (size_t)(nt * 16 + l16) * 64 + kc * 32, half);
        acc[nt] = wmma_bf16(a, bb, acc[nt]);
      }
    }
#pragma unroll
    for (int nt = 0; nt < 6; nt++)
#pragma unroll
      for (int j = 0; j < 8; j++) {
        int t = wv * 16 + j + half * 8;
        int o = nt * 16 + l16;
        Om[t * CPAD + o] = (bf16)acc[nt][j];
      }
  }
  __syncthreads();

  // ---- out-proj + residual + LayerNorm (cross-lane butterfly) + store
  {
    v8f acc[6];
#pragma unroll
    for (int i = 0; i < 6; i++) acc[i] = vzero8();
    for (int kc = 0; kc < 3; kc++) {
      v16bf a = frag_a(Om + (wv * 16 + l16) * CPAD + kc * 32, half);
#pragma unroll
      for (int nt = 0; nt < 6; nt++) {
        v16bf bb = frag_b(Wo + (size_t)(nt * 16 + l16) * CPAD + kc * 32, half);
        acc[nt] = wmma_bf16(a, bb, acc[nt]);
      }
    }
#pragma unroll
    for (int j = 0; j < 8; j++) {
      int t = wv * 16 + j + half * 8;
      float yv[6];
      float s1 = 0.f, s2 = 0.f;
#pragma unroll
      for (int nt = 0; nt < 6; nt++) {
        int c = nt * 16 + l16;
        float v = acc[nt][j];
        if (c < 67)
          v += bo[c] + (float)Xp[t * CPAD + c];
        else
          v = 0.f;
        yv[nt] = v;
        s1 += v;
        s2 += v * v;
      }
#pragma unroll
      for (int m = 1; m < 16; m <<= 1) {
        s1 += __shfl_xor(s1, m, 32);
        s2 += __shfl_xor(s2, m, 32);
      }
      float mu = s1 * (1.f / 67.f);
      float var = s2 * (1.f / 67.f) - mu * mu;
      float rstd = rsqrtf(var + 1e-5f);
#pragma unroll
      for (int nt = 0; nt < 6; nt++) {
        int c = nt * 16 + l16;
        bf16 ov = (bf16)0.f;
        if (c < 67) ov = (bf16)(lng[c] * (yv[nt] - mu) * rstd + lnb[c]);
        Yout[(size_t)c * M2COLS + colbase + t] = ov;  // pad rows zeroed for f1
      }
    }
  }
}

// ================= fused f1 -> f2 -> max-pool-over-T =================
__global__ __launch_bounds__(256) void f1f2_max_kernel(
    const bf16* __restrict__ Y, const bf16* __restrict__ F1p,
    const bf16* __restrict__ F2p, const float* __restrict__ s1,
    const float* __restrict__ h1, const float* __restrict__ s2,
    const float* __restrict__ h2s, float* __restrict__ out) {
  __shared__ alignas(16) bf16 Bs[2][128 * 40];
  __shared__ alignas(16) bf16 Ht[128 * 136];  // transposed intermediate [col][ch]
  const int tid = threadIdx.x;
  const int lane = tid & 31, wv = tid >> 5;
  const int half = (lane >> 4) & 1, l16 = lane & 15;
  const int wr = wv >> 1, wc = wv & 1;
  const int tileM = blockIdx.x * 128;

  auto stageY = [&](int buf, int k0) {
    int kk = tid >> 3, c0 = (tid & 7) * 16;
    const bf16* src = Y + (size_t)(k0 + kk) * M2COLS + tileM + c0;
    v8bf v0 = *(const v8bf*)(src);
    v8bf v1 = *(const v8bf*)(src + 8);
#pragma unroll
    for (int j = 0; j < 8; j++) {
      Bs[buf][(c0 + j) * 40 + kk] = v0[j];
      Bs[buf][(c0 + 8 + j) * 40 + kk] = v1[j];
    }
  };

  // ---- stage 1: H = relu(bn1(F1p @ Y[:, tile])), K = 96, double-buffered
  v8f acc[2][4];
#pragma unroll
  for (int a = 0; a < 2; a++)
#pragma unroll
    for (int c = 0; c < 4; c++) acc[a][c] = vzero8();
  stageY(0, 0);
  __syncthreads();
  for (int kt = 0; kt < 3; kt++) {
    int cur = kt & 1;
    if (kt + 1 < 3) stageY(cur ^ 1, (kt + 1) * 32);
#pragma unroll
    for (int mi = 0; mi < 2; mi++) {
      v16bf a = frag_a(F1p + (size_t)(wr * 32 + mi * 16 + l16) * CPAD + kt * 32, half);
#pragma unroll
      for (int ni = 0; ni < 4; ni++) {
        v16bf bb = frag_b(&Bs[cur][(wc * 64 + ni * 16 + l16) * 40], half);
        acc[mi][ni] = wmma_bf16(a, bb, acc[mi][ni]);
      }
    }
    __syncthreads();
  }
#pragma unroll
  for (int mi = 0; mi < 2; mi++)
#pragma unroll
    for (int ni = 0; ni < 4; ni++)
#pragma unroll
      for (int j = 0; j < 8; j++) {
        int o = wr * 32 + mi * 16 + j + half * 8;
        int lcol = wc * 64 + ni * 16 + l16;
        float y = s1[o] * acc[mi][ni][j] + h1[o];
        y = y > 0.f ? y : 0.f;
        Ht[lcol * 136 + o] = (bf16)y;
      }
  __syncthreads();
  // ---- stage 2: relu(bn2(F2p @ H)) then max over T(=64-col groups)
  v8f acc2[2][4];
#pragma unroll
  for (int a = 0; a < 2; a++)
#pragma unroll
    for (int c = 0; c < 4; c++) acc2[a][c] = vzero8();
  for (int k0 = 0; k0 < 128; k0 += 32) {
#pragma unroll
    for (int mi = 0; mi < 2; mi++) {
      v16bf a = frag_a(F2p + (size_t)(wr * 32 + mi * 16 + l16) * 128 + k0, half);
#pragma unroll
      for (int ni = 0; ni < 4; ni++) {
        v16bf bb = frag_b(Ht + (wc * 64 + ni * 16 + l16) * 136 + k0, half);
        acc2[mi][ni] = wmma_bf16(a, bb, acc2[mi][ni]);
      }
    }
  }
  int gid = blockIdx.x * 2 + wc;  // = bh*512 + n
  int bh = gid >> 9, n = gid & 511;
#pragma unroll
  for (int mi = 0; mi < 2; mi++)
#pragma unroll
    for (int j = 0; j < 8; j++) {
      int o = wr * 32 + mi * 16 + j + half * 8;
      float vmax = -1e30f;
#pragma unroll
      for (int ni = 0; ni < 4; ni++) {
        float y = s2[o] * acc2[mi][ni][j] + h2s[o];
        y = y > 0.f ? y : 0.f;
        vmax = fmaxf(vmax, y);
      }
#pragma unroll
      for (int m = 1; m < 16; m <<= 1) vmax = fmaxf(vmax, __shfl_xor(vmax, m, 32));
      if (l16 == 0) out[((size_t)bh * 128 + o) * NPTS + n] = vmax;
    }
}

// ================= gp transpose tail output =================
__global__ void gp_out_kernel(const float* __restrict__ gp_f, float* __restrict__ out) {
  int i = blockIdx.x * 256 + threadIdx.x;
  if (i >= 2 * NPTS * 3) return;
  int b = i / (NPTS * 3), rem = i % (NPTS * 3), n = rem / 3, c = rem % 3;
  out[i] = gp_f[(b * 3 + c) * NPTS + n];
}

// ================= host launcher =================
extern "C" void kernel_launch(void* const* d_in, const int* in_sizes, int n_in,
                              void* d_out, int out_size, void* d_ws, size_t ws_size,
                              hipStream_t stream) {
  (void)in_sizes; (void)n_in; (void)out_size; (void)ws_size;
  const float* gp_f   = (const float*)d_in[0];
  const float* gpfeat = (const float*)d_in[1];
  const float* rp_f   = (const float*)d_in[2];
  const float* rpfeat = (const float*)d_in[3];
  const float* vprot  = (const float*)d_in[4];
  const float* m1_w = (const float*)d_in[5],  *m1_b = (const float*)d_in[6];
  const float* m1_g = (const float*)d_in[7],  *m1_be = (const float*)d_in[8];
  const float* m1_mu = (const float*)d_in[9], *m1_v = (const float*)d_in[10];
  const float* m2_w = (const float*)d_in[11], *m2_b = (const float*)d_in[12];
  const float* m2_g = (const float*)d_in[13], *m2_be = (const float*)d_in[14];
  const float* m2_mu = (const float*)d_in[15], *m2_v = (const float*)d_in[16];
  const float* awq = (const float*)d_in[17], *abq = (const float*)d_in[18];
  const float* awk = (const float*)d_in[19], *abk = (const float*)d_in[20];
  const float* awv = (const float*)d_in[21], *abv = (const float*)d_in[22];
  const float* awo = (const float*)d_in[23], *abo = (const float*)d_in[24];
  const float* lng = (const float*)d_in[25], *lnb = (const float*)d_in[26];
  const float* f1_w = (const float*)d_in[27], *f1_b = (const float*)d_in[28];
  const float* f1_g = (const float*)d_in[29], *f1_be = (const float*)d_in[30];
  const float* f1_mu = (const float*)d_in[31], *f1_v = (const float*)d_in[32];
  const float* f2_w = (const float*)d_in[33], *f2_b = (const float*)d_in[34];
  const float* f2_g = (const float*)d_in[35], *f2_be = (const float*)d_in[36];
  const float* f2_mu = (const float*)d_in[37], *f2_v = (const float*)d_in[38];
  float* out = (float*)d_out;

  char* ws = (char*)d_ws;
  int*   idx = (int*)(ws + OFF_IDX);
  float* xyz = (float*)(ws + OFF_XYZ);
  float* s1 = (float*)(ws + OFF_S1),  *sh1 = (float*)(ws + OFF_SH1);
  float* s2 = (float*)(ws + OFF_S2),  *sh2 = (float*)(ws + OFF_SH2);
  float* sf1 = (float*)(ws + OFF_SF1), *shf1 = (float*)(ws + OFF_SHF1);
  float* sf2 = (float*)(ws + OFF_SF2), *shf2 = (float*)(ws + OFF_SHF2);
  bf16* WQp = (bf16*)(ws + OFF_WQ), *WKp = (bf16*)(ws + OFF_WK);
  bf16* WVp = (bf16*)(ws + OFF_WV), *WOp = (bf16*)(ws + OFF_WO);
  bf16* W1p = (bf16*)(ws + OFF_W1P), *W2p = (bf16*)(ws + OFF_W2P);
  bf16* F1p = (bf16*)(ws + OFF_F1P), *F2p = (bf16*)(ws + OFF_F2P);
  bf16* H1 = (bf16*)(ws + OFF_H1), *H2 = (bf16*)(ws + OFF_H2);
  bf16* X1 = (bf16*)(ws + OFF_X1), *Y = (bf16*)(ws + OFF_Y);

  // ---- packing / query / gp tail
  pack_w_kernel<<<(256 * K1PAD + 255) / 256, 256, 0, stream>>>(m1_w, W1p, 256, 515, 256, K1PAD);
  pack_w_kernel<<<(768 * 256 + 255) / 256, 256, 0, stream>>>(m2_w, W2p, 768, 256, 768, 256);
  pack_w_kernel<<<(128 * CPAD + 255) / 256, 256, 0, stream>>>(f1_w, F1p, 128, 67, 128, CPAD);
  pack_w_kernel<<<(128 * 128 + 255) / 256, 256, 0, stream>>>(f2_w, F2p, 128, 128, 128, 128);
  pack_w_kernel<<<(CPAD * CPAD + 255) / 256, 256, 0, stream>>>(awq, WQp, 67, 67, CPAD, CPAD);
  pack_w_kernel<<<(CPAD * CPAD + 255) / 256, 256, 0, stream>>>(awk, WKp, 67, 67, CPAD, CPAD);
  pack_w_kernel<<<(CPAD * CPAD + 255) / 256, 256, 0, stream>>>(awv, WVp, 67, 67, CPAD, CPAD);
  pack_w_kernel<<<(CPAD * CPAD + 255) / 256, 256, 0, stream>>>(awo, WOp, 67, 67, CPAD, CPAD);
  pack_bn_kernel<<<1, 256, 0, stream>>>(m1_b, m1_g, m1_be, m1_mu, m1_v, s1, sh1, 256);
  pack_bn_kernel<<<3, 256, 0, stream>>>(m2_b, m2_g, m2_be, m2_mu, m2_v, s2, sh2, 768);
  pack_bn_kernel<<<1, 256, 0, stream>>>(f1_b, f1_g, f1_be, f1_mu, f1_v, sf1, shf1, 128);
  pack_bn_kernel<<<1, 256, 0, stream>>>(f2_b, f2_g, f2_be, f2_mu, f2_v, sf2, shf2, 128);
  cylinder_query_kernel<<<4, 256, 0, stream>>>(gp_f, vprot, idx);
  gp_out_kernel<<<12, 256, 0, stream>>>(gp_f, out + (size_t)24 * 128 * NPTS);

  // ---- feature build
  build_feats_kernel<<<2 * NPTS, 256, 0, stream>>>(gp_f, gpfeat, rp_f, rpfeat, vprot,
                                                   idx, X1, xyz);

  // ---- SharedMLP GEMMs (bf16 WMMA, double-buffered)
  gemm_bn_relu_kernel<<<dim3(M1COLS / 128, 2), 256, 0, stream>>>(W1p, X1, H1, s1, sh1,
                                                                 M1COLS, K1PAD);
  gemm_bn_relu_kernel<<<dim3(M1COLS / 128, 6), 256, 0, stream>>>(W2p, H1, H2, s2, sh2,
                                                                 M1COLS, 256);

  // ---- fused self-attention + layernorm (X1 dead -> Y overlays it)
  attn_kernel<<<24 * NPTS, 128, 0, stream>>>(H2, xyz, WQp, WKp, WVp, WOp, abq, abk,
                                             abv, abo, lng, lnb, Y);

  // ---- fused f1 -> f2 -> max-pool, writes final output
  f1f2_max_kernel<<<M2COLS / 128, 256, 0, stream>>>(Y, F1p, F2p, sf1, shf1, sf2, shf2,
                                                    out);
}
